// SpikingSelfAttention_87969520156922
// MI455X (gfx1250) — 16-bit matrix operand order repaired and checked by reading, not compiled, not run on hardware
//
#include <hip/hip_runtime.h>

typedef __attribute__((ext_vector_type(16))) _Float16      v16h;
typedef __attribute__((ext_vector_type(8)))  _Float16      v8h;
typedef __attribute__((ext_vector_type(8)))  float         v8f;
typedef __attribute__((ext_vector_type(8)))  int           v8i;

constexpr int    TT = 2, BB = 16, CC = 256, NN = 4096, NH = 8;
constexpr size_t SEG = (size_t)TT * BB * CC * NN;
constexpr size_t BCN = (size_t)BB * CC * NN;

constexpr size_t OFF_W   = 0;
constexpr size_t OFF_BN  = 524288;
constexpr size_t OFF_KVT = 532480;
constexpr size_t OFF_XS  = 1056768;
constexpr size_t OFF_QF  = OFF_XS + SEG * 2;
constexpr size_t OFF_KU  = OFF_QF + SEG * 2;
constexpr size_t OFF_VU  = OFF_KU + SEG;
constexpr size_t OFF_SF  = OFF_XS;

__global__ void prep_weights_shuf(const float* __restrict__ wq, const float* __restrict__ wk,
                                  const float* __restrict__ wv, const float* __restrict__ wp,
                                  _Float16* __restrict__ w16) {
  int idx  = blockIdx.x * 256 + threadIdx.x;
  int lane = idx & 31;
  int ks   = (idx >> 5) & 7;
  int ot   = (idx >> 8) & 15;
  int mat  = idx >> 12;
  const float* W = (mat == 0) ? wq : (mat == 1) ? wk : (mat == 2) ? wv : wp;
  int m = lane & 15, kh = lane >> 4;
  v16h frag;
#pragma unroll
  for (int j = 0; j < 16; ++j) {
    int K = ks * 32 + ((j < 8) ? (kh * 8 + j) : (16 + kh * 8 + (j - 8)));
    frag[j] = (_Float16)W[(ot * 16 + m) * CC + K];
  }
  *(v16h*)(w16 + (size_t)idx * 16) = frag;
}

__global__ void prep_bn(const float* qg, const float* qb, const float* qm, const float* qv,
                        const float* kg, const float* kb, const float* km, const float* kv_,
                        const float* vg, const float* vb, const float* vm, const float* vv,
                        const float* pg, const float* pb, const float* pm, const float* pv,
                        const float* bp, float* __restrict__ o) {
  int c = threadIdx.x;
  float i0 = qg[c] * rsqrtf(qv[c] + 1e-5f);  o[c]        = i0; o[256  + c] = qb[c] - qm[c] * i0;
  float i1 = kg[c] * rsqrtf(kv_[c] + 1e-5f); o[512  + c] = i1; o[768  + c] = kb[c] - km[c] * i1;
  float i2 = vg[c] * rsqrtf(vv[c] + 1e-5f);  o[1024 + c] = i2; o[1280 + c] = vb[c] - vm[c] * i2;
  float i3 = pg[c] * rsqrtf(pv[c] + 1e-5f);  o[1536 + c] = i3;
  o[1792 + c] = bp[c] * i3 + (pb[c] - pm[c] * i3);
}

__global__ void lif_input(const float* __restrict__ x, float* __restrict__ xsF,
                          _Float16* __restrict__ xsH) {
  size_t i = (size_t)blockIdx.x * 256 + threadIdx.x;
  float x0 = x[i], x1 = x[BCN + i];
  float v0 = x0 * 0.5f;
  bool  s0 = v0 >= 1.0f;
  float v0r = s0 ? 0.0f : v0;
  float v1 = (v0r + x1) * 0.5f;
  bool  s1 = v1 >= 1.0f;
  xsF[i]       = s0 ? 1.0f : 0.0f;
  xsF[BCN + i] = s1 ? 1.0f : 0.0f;
  xsH[i]       = s0 ? (_Float16)1.0f : (_Float16)0.0f;
  xsH[BCN + i] = s1 ? (_Float16)1.0f : (_Float16)0.0f;
}

template <int AUX>
__global__ void __launch_bounds__(256) spike_gemm(
    const _Float16* __restrict__ Ashuf,
    const _Float16* __restrict__ X,
    const float* __restrict__ scale, const float* __restrict__ shift,
    float* __restrict__ outF, void* __restrict__ aux, float vth) {
  int lane = threadIdx.x & 31, wave = threadIdx.x >> 5;
  int b   = blockIdx.x >> 8;
  int rem = blockIdx.x & 255;
  int op  = rem >> 5;
  int ntg = rem & 31;
  int o0  = op * 32;
  int n0  = (ntg * 8 + wave) * 16;
  int m = lane & 15, kh = lane >> 4;

  __shared__ v16h lA[512];
  {
    const v16h* gA = (const v16h*)(Ashuf + (size_t)(op * 2) * 8 * 32 * 16);
    int t2 = threadIdx.x * 2;
    lA[t2]     = gA[t2];
    lA[t2 + 1] = gA[t2 + 1];
  }
  __syncthreads();

  v8f acc[2][2] = {};
  const _Float16* x0p = X + ((size_t)(0 * BB + b) * CC) * NN + n0 + m;
  const _Float16* x1p = X + ((size_t)(1 * BB + b) * CC) * NN + n0 + m;

  for (int ks = 0; ks < 8; ++ks) {
    size_t kofs = (size_t)(ks * 32) * NN;
    v16h a0 = lA[(0 * 8 + ks) * 32 + lane];
    v16h a1 = lA[(1 * 8 + ks) * 32 + lane];
    v16h b0, b1;
#pragma unroll
    for (int j = 0; j < 16; ++j) {
      size_t kk = kofs + (size_t)((j < 8) ? (kh * 8 + j) : (16 + kh * 8 + (j - 8))) * NN;
      b0[j] = x0p[kk];
      b1[j] = x1p[kk];
    }
    if (ks < 7) {
      __builtin_prefetch(x0p + kofs + (size_t)32 * NN, 0, 3);
      __builtin_prefetch(x1p + kofs + (size_t)32 * NN, 0, 3);
    }
    acc[0][0] = __builtin_amdgcn_wmma_f32_16x16x32_f16(false, a0, false, b0, (short)0, acc[0][0], false, false);
    acc[0][1] = __builtin_amdgcn_wmma_f32_16x16x32_f16(false, a1, false, b0, (short)0, acc[0][1], false, false);
    acc[1][0] = __builtin_amdgcn_wmma_f32_16x16x32_f16(false, a0, false, b1, (short)0, acc[1][0], false, false);
    acc[1][1] = __builtin_amdgcn_wmma_f32_16x16x32_f16(false, a1, false, b1, (short)0, acc[1][1], false, false);
  }

#pragma unroll
  for (int iot = 0; iot < 2; ++iot) {
#pragma unroll
    for (int r = 0; r < 8; ++r) {
      int o = o0 + iot * 16 + r + kh * 8;
      int n = n0 + m;
      float sc = scale[o], sh = shift[o];
      float y0 = acc[0][iot][r] * sc + sh;
      float y1 = acc[1][iot][r] * sc + sh;
      size_t i0 = ((size_t)(0 * BB + b) * CC + o) * NN + n;
      size_t i1 = ((size_t)(1 * BB + b) * CC + o) * NN + n;
      if (AUX != 0) {
        float v0 = y0 * 0.5f;
        bool  s0 = v0 >= vth;
        float v0r = s0 ? 0.0f : v0;
        float v1 = (v0r + y1) * 0.5f;
        bool  s1 = v1 >= vth;
        outF[i0] = s0 ? 1.0f : 0.0f;
        outF[i1] = s1 ? 1.0f : 0.0f;
        if (AUX == 1) {
          ((_Float16*)aux)[i0] = s0 ? (_Float16)1.0f : (_Float16)0.0f;
          ((_Float16*)aux)[i1] = s1 ? (_Float16)1.0f : (_Float16)0.0f;
        } else {
          ((unsigned char*)aux)[i0] = s0 ? 1 : 0;
          ((unsigned char*)aux)[i1] = s1 ? 1 : 0;
        }
      } else {
        outF[i0] = y0;
        outF[i1] = y1;
      }
    }
  }
}

__global__ void kv_gemm(const unsigned char* __restrict__ K8,
                        const unsigned char* __restrict__ V8,
                        _Float16* __restrict__ kvT) {
  int lane = threadIdx.x & 31, wave = threadIdx.x >> 5;
  int gw   = blockIdx.x * 8 + wave;
  int ctx  = gw >> 2;
  int tile = gw & 3;
  int d0 = (tile >> 1) * 16, e0 = (tile & 1) * 16;
  int t = ctx >> 7, b = (ctx >> 3) & 15, h = ctx & 7;
  int m = lane & 15, half = lane >> 4;

  const unsigned char* ka = K8 + ((size_t)(t * BB + b) * CC + h * 32 + d0 + m) * NN + half * 32;
  const unsigned char* vb = V8 + ((size_t)(t * BB + b) * CC + h * 32 + e0 + m) * NN + half * 32;

  v8i acc = {};
  for (int ns = 0; ns < 64; ++ns) {
    v8i araw = *(const v8i*)(ka + (size_t)ns * 64);
    v8i braw = *(const v8i*)(vb + (size_t)ns * 64);
    acc = __builtin_amdgcn_wmma_i32_16x16x64_iu8(false, araw, false, braw, acc, false, false);
  }
  _Float16* dst = kvT + (size_t)ctx * 1024;
#pragma unroll
  for (int r = 0; r < 8; ++r) {
    int d = d0 + r + half * 8;
    int e = e0 + m;
    dst[e * 32 + d] = (_Float16)(float)acc[r];
  }
}

__global__ void attn_lif(const _Float16* __restrict__ Q,
                         const _Float16* __restrict__ kvT,
                         _Float16* __restrict__ S) {
  int lane = threadIdx.x & 31, wave = threadIdx.x >> 5;
  int gw  = blockIdx.x * 8 + wave;
  int b   = gw >> 11;
  int rem = gw & 2047;
  int h   = rem >> 8;
  int n0  = (rem & 255) << 4;
  int m = lane & 15, kh = lane >> 4;

#pragma unroll
  for (int eh = 0; eh < 2; ++eh) {
    int e0 = eh * 16;
    v8f accT[2];
#pragma unroll
    for (int t = 0; t < 2; ++t) {
      const _Float16* arow = kvT + (size_t)((t * BB + b) * NH + h) * 1024 + (size_t)(e0 + m) * 32;
      v8h alo = *(const v8h*)(arow + kh * 8);
      v8h ahi = *(const v8h*)(arow + 16 + kh * 8);
      v16h a;
#pragma unroll
      for (int j = 0; j < 8; ++j) { a[j] = alo[j]; a[j + 8] = ahi[j]; }
      const _Float16* qp = Q + ((size_t)(t * BB + b) * CC + h * 32) * NN + n0 + m;
      v16h bm;
#pragma unroll
      for (int j = 0; j < 16; ++j)
        bm[j] = qp[(size_t)((j < 8) ? (kh * 8 + j) : (16 + kh * 8 + (j - 8))) * NN];
      v8f z = {};
      accT[t] = __builtin_amdgcn_wmma_f32_16x16x32_f16(false, a, false, bm, (short)0, z, false, false);
    }
#pragma unroll
    for (int r = 0; r < 8; ++r) {
      int e = e0 + r + kh * 8;
      int n = n0 + m;
      float u0 = accT[0][r] * 0.125f;
      float u1 = accT[1][r] * 0.125f;
      float v0 = u0 * 0.5f;
      bool  s0 = v0 >= 0.5f;
      float v0r = s0 ? 0.0f : v0;
      float v1 = (v0r + u1) * 0.5f;
      bool  s1 = v1 >= 0.5f;
      size_t c = (size_t)h * 32 + e;
      S[((size_t)(0 * BB + b) * CC + c) * NN + n] = s0 ? (_Float16)1.0f : (_Float16)0.0f;
      S[((size_t)(1 * BB + b) * CC + c) * NN + n] = s1 ? (_Float16)1.0f : (_Float16)0.0f;
    }
  }
}

extern "C" void kernel_launch(void* const* d_in, const int* in_sizes, int n_in,
                              void* d_out, int out_size, void* d_ws, size_t ws_size,
                              hipStream_t stream) {
  const float* x  = (const float*)d_in[0];
  const float* wq = (const float*)d_in[1];
  const float* wk = (const float*)d_in[2];
  const float* wv = (const float*)d_in[3];
  const float* wp = (const float*)d_in[4];
  const float* bp = (const float*)d_in[5];

  char* ws = (char*)d_ws;
  _Float16*      W16 = (_Float16*)(ws + OFF_W);
  float*         BN  = (float*)(ws + OFF_BN);
  _Float16*      KVT = (_Float16*)(ws + OFF_KVT);
  _Float16*      XS  = (_Float16*)(ws + OFF_XS);
  _Float16*      QF  = (_Float16*)(ws + OFF_QF);
  unsigned char* KU  = (unsigned char*)(ws + OFF_KU);
  unsigned char* VU  = (unsigned char*)(ws + OFF_VU);
  _Float16*      SF  = (_Float16*)(ws + OFF_SF);

  float* outF = (float*)d_out;

  prep_weights_shuf<<<64, 256, 0, stream>>>(wq, wk, wv, wp, W16);
  prep_bn<<<1, 256, 0, stream>>>((const float*)d_in[6],  (const float*)d_in[7],
                                 (const float*)d_in[8],  (const float*)d_in[9],
                                 (const float*)d_in[10], (const float*)d_in[11],
                                 (const float*)d_in[12], (const float*)d_in[13],
                                 (const float*)d_in[14], (const float*)d_in[15],
                                 (const float*)d_in[16], (const float*)d_in[17],
                                 (const float*)d_in[18], (const float*)d_in[19],
                                 (const float*)d_in[20], (const float*)d_in[21],
                                 bp, BN);
  lif_input<<<65536, 256, 0, stream>>>(x, outF + SEG, XS);

  spike_gemm<1><<<4096, 256, 0, stream>>>(W16,          XS, BN,        BN + 256,  outF + 2 * SEG, QF, 1.0f);
  spike_gemm<2><<<4096, 256, 0, stream>>>(W16 + 65536,  XS, BN + 512,  BN + 768,  outF + 3 * SEG, KU, 1.0f);
  spike_gemm<2><<<4096, 256, 0, stream>>>(W16 + 131072, XS, BN + 1024, BN + 1280, outF + 4 * SEG, VU, 1.0f);

  kv_gemm<<<128, 256, 0, stream>>>(KU, VU, KVT);
  attn_lif<<<4096, 256, 0, stream>>>(QF, KVT, SF);

  spike_gemm<0><<<4096, 256, 0, stream>>>(W16 + 196608, SF, BN + 1536, BN + 1792, outF, nullptr, 0.0f);
}
